// MeshAttention_26843545600329
// MI455X (gfx1250) — matrix operand order repaired and checked by reading, not compiled, not run on hardware
//
#include <hip/hip_runtime.h>

// ---------------------------------------------------------------------------
// MeshAttention for MI455X (gfx1250): wave32 + v_wmma_f32_16x16x32_f16.
// Fused scores->softmax->A*V kernel keeps the 16 x 2048 attention slab in the
// WGP's 320KB LDS (CDNA5-only: 128KB static LDS), so the 134MB `a` matrix is
// written to HBM exactly once and never re-read.
// B=2, D=256, E=2048, H=4, DK=DV=64, MAX_POS=8, ATTN_MAX_DIST=8
// ---------------------------------------------------------------------------

#define B_   2
#define D_   256
#define E_   2048
#define H_   4
#define DK_  64
#define HD_  (H_ * DK_)   // 256
#define LDST 2052         // LDS row stride (floats): 2048 + 4 -> conflict-free

typedef _Float16 half8 __attribute__((ext_vector_type(8)));
typedef _Float16 v16h  __attribute__((ext_vector_type(16)));
typedef float    v8f   __attribute__((ext_vector_type(8)));
typedef float    f4    __attribute__((ext_vector_type(4)));

// ---- WMMA helpers ----------------------------------------------------------

__device__ __forceinline__ v8f wmma_f16(v16h a, v16h b, v8f c) {
  // D = A(16x32 f16) * B(32x16 f16) + C(16x16 f32)
  return __builtin_amdgcn_wmma_f32_16x16x32_f16(
      /*neg_a=*/false, a, /*neg_b=*/false, b,
      /*c_mod=*/(short)0, c, /*reuse_a=*/false, /*reuse_b=*/false);
}

// A-matrix 16x32 f16 fragment (row-major source, ld in halves).
// lane 0-15: M=lane, K quadrants {0..7, 16..23}; lanes 16-31: {8..15, 24..31}
__device__ __forceinline__ v16h load_a16(const _Float16* base, int ld, int lane) {
  int m = lane & 15;
  int koff = (lane >> 4) * 8;
  const half8* p0 = (const half8*)(base + (size_t)m * ld + koff);
  const half8* p1 = (const half8*)(base + (size_t)m * ld + koff + 16);
  half8 lo = *p0, hi = *p1;
  v16h r;
#pragma unroll
  for (int i = 0; i < 8; ++i) { r[i] = lo[i]; r[i + 8] = hi[i]; }
  return r;
}

// B-matrix 32x16 f16 fragment from a row-major [N, K] source (i.e. B^T):
// lanes 0-15: K={0..7,16..23} at column N=lane; lanes 16-31: K={8..15,24..31}.
__device__ __forceinline__ v16h load_b16(const _Float16* base, int ld, int lane) {
  int n = lane & 15;
  int koff = (lane >> 4) * 8;
  const half8* p0 = (const half8*)(base + (size_t)n * ld + koff);
  const half8* p1 = (const half8*)(base + (size_t)n * ld + koff + 16);
  half8 lo = *p0, hi = *p1;
  v16h r;
#pragma unroll
  for (int i = 0; i < 8; ++i) { r[i] = lo[i]; r[i + 8] = hi[i]; }
  return r;
}

// A-matrix fragment from an f32 source (LDS or global), converted on load.
__device__ __forceinline__ v16h load_a32(const float* base, int ld, int lane) {
  int m = lane & 15;
  int koff = (lane >> 4) * 8;
  const float* p = base + (size_t)m * ld + koff;
  f4 x0 = *(const f4*)(p);
  f4 x1 = *(const f4*)(p + 4);
  f4 x2 = *(const f4*)(p + 16);
  f4 x3 = *(const f4*)(p + 20);
  v16h r;
#pragma unroll
  for (int i = 0; i < 4; ++i) {
    r[i]      = (_Float16)x0[i];
    r[4 + i]  = (_Float16)x1[i];
    r[8 + i]  = (_Float16)x2[i];
    r[12 + i] = (_Float16)x3[i];
  }
  return r;
}

// ---- Kernel 1: transpose x -> s, layernorm -> qn (both f16 [B,E,D]) --------

__global__ void ln_kernel(const float* __restrict__ x, const float* __restrict__ g,
                          const float* __restrict__ be, _Float16* __restrict__ s_h,
                          _Float16* __restrict__ qn_h) {
  __shared__ float tile[32][D_ + 1];  // padded stride kills bank conflicts
  int blk = blockIdx.x;                       // B * E/32 blocks
  int b = blk / (E_ / 32);
  int ebase = (blk % (E_ / 32)) * 32;
  int tid = threadIdx.x;
  int eoff = tid & 31, dgrp = tid >> 5;       // 8 waves, one d-slice each

  // coalesced load of a 256(d) x 32(e) slab of x[b, :, ebase..]
#pragma unroll 4
  for (int dp = 0; dp < 32; ++dp) {
    int d = dp * 8 + dgrp;
    tile[eoff][d] = x[((size_t)b * D_ + d) * E_ + ebase + eoff];
  }
  __syncthreads();

  int e = tid >> 3, sub = tid & 7;            // 8 threads cooperate per e-row
  float sum = 0.f, sq = 0.f;
#pragma unroll 8
  for (int j = 0; j < 32; ++j) {
    float v = tile[e][sub + j * 8];
    sum += v; sq += v * v;
  }
#pragma unroll
  for (int o = 4; o >= 1; o >>= 1) {          // reduce across the 8 sub-lanes
    sum += __shfl_xor(sum, o, 32);
    sq  += __shfl_xor(sq,  o, 32);
  }
  float mu   = sum * (1.0f / D_);
  float var  = sq * (1.0f / D_) - mu * mu;
  float rstd = rsqrtf(var + 1e-6f);

  size_t rowo = ((size_t)b * E_ + ebase + e) * D_;
#pragma unroll 8
  for (int j = 0; j < 32; ++j) {
    int d = sub + j * 8;
    float v = tile[e][d];
    s_h[rowo + d]  = (_Float16)v;
    qn_h[rowo + d] = (_Float16)((v - mu) * rstd * g[d] + be[d]);
  }
}

// ---- Kernel 2: convert the four 256x256 weight matrices to f16 -------------

__global__ void wcvt_kernel(const float* __restrict__ a, const float* __restrict__ b,
                            const float* __restrict__ c, const float* __restrict__ d,
                            _Float16* __restrict__ out) {
  int i = blockIdx.x * 256 + threadIdx.x;     // 4 * 65536 elements
  int which = i >> 16, j = i & 65535;
  const float* src = (which == 0) ? a : (which == 1) ? b : (which == 2) ? c : d;
  out[i] = (_Float16)src[j];
}

// ---- Kernel 3: q/k/v projections (WMMA GEMM [B*E,256] x [256,256]^T) -------
// which (blockIdx.z): 0 = q (from qn, scaled by dk^-1/2), 1 = k, 2 = v (stored
// transposed [B,H,DV,E] so it is a row-major WMMA B-operand for A*V).

__global__ void proj_kernel(const _Float16* __restrict__ s_h, const _Float16* __restrict__ qn_h,
                            const _Float16* __restrict__ w_all,
                            _Float16* __restrict__ qf, _Float16* __restrict__ kf,
                            _Float16* __restrict__ vT) {
  int which = blockIdx.z;
  const _Float16* A = (which == 0) ? qn_h : s_h;
  const _Float16* W = w_all + (size_t)which * 65536;
  int lane = threadIdx.x & 31;
  int wave = threadIdx.x >> 5;
  int mbase = blockIdx.y * 128 + wave * 16;   // row in [B*E]
  int nbase = blockIdx.x * 16;                // out feature
  v8f c = {};
#pragma unroll
  for (int kb = 0; kb < 256; kb += 32) {
    v16h af = load_a16(A + (size_t)mbase * 256 + kb, 256, lane);
    v16h bf = load_b16(W + (size_t)nbase * 256 + kb, 256, lane);
    c = wmma_f16(af, bf, c);
  }
  int l16 = lane & 15, hfl = lane >> 4;
  int n = nbase + l16;
  int h = n >> 6, d = n & 63;
  float scale = (which == 0) ? 0.125f : 1.0f; // 1/sqrt(dk)
  int b = mbase / E_;
#pragma unroll
  for (int r = 0; r < 8; ++r) {
    int e = mbase - b * E_ + r + 8 * hfl;
    _Float16 val = (_Float16)(c[r] * scale);
    if (which == 2) {
      vT[(((size_t)b * H_ + h) * DK_ + d) * E_ + e] = val;
    } else {
      _Float16* dst = (which == 0) ? qf : kf;
      dst[(((size_t)b * H_ + h) * E_ + e) * DK_ + d] = val;
    }
  }
}

// ---- Kernel 4: q_dot_rpr[b,h,e,p] = qs . w_rpr[p]  (p padded to 16) --------

__global__ void qrpr_kernel(const _Float16* __restrict__ qf, const float* __restrict__ w_rpr,
                            float* __restrict__ qrpr) {
  int i = blockIdx.x * 256 + threadIdx.x;     // B*H*E*16
  int row = i >> 4, p = i & 15;
  float acc = 0.f;
  if (p <= 8) {
    const _Float16* qp = qf + (size_t)row * DK_;
    const float* wp = w_rpr + p * DK_;
#pragma unroll 8
    for (int j = 0; j < DK_; ++j) acc += (float)qp[j] * wp[j];
  }
  qrpr[i] = acc;
}

// ---- Kernel 5 (fused): scores + bias/mask + softmax + A*V ------------------
// One block = 16 query rows x all 2048 keys of one (b,h), slab held in LDS.
//  Phase 1: 8 waves x (16x256) WMMA logit strips -> LDS.
//  Phase 2: per-row max / exp / sum in LDS; keep 1/sum (normalization folded
//           into the consumers, no third LDS pass).
//  Phase 3: waves 0-3 compute A*V from LDS via WMMA (f32->f16 on ds loads);
//           waves 4-7 concurrently stream normalized `a` to HBM (coalesced).

__global__ void __launch_bounds__(256)
attn_fused_kernel(const _Float16* __restrict__ qf, const _Float16* __restrict__ kf,
                  const float* __restrict__ qrpr, const int* __restrict__ dist,
                  const _Float16* __restrict__ vT,
                  float* __restrict__ aout, _Float16* __restrict__ outcat) {
  __shared__ float tile[16 * LDST];   // ~128.3 KB (CDNA5 WGP has 320 KB)
  __shared__ float inv_s[16];
  int bh = blockIdx.y;                // 0..B*H-1
  int b = bh >> 2, h = bh & 3;        // H_ == 4
  int qbase = blockIdx.x * 16;
  int tid = threadIdx.x;
  int lane = tid & 31, wave = tid >> 5;
  int l16 = lane & 15, hfl = lane >> 4;

  // -------- Phase 1: masked, biased logits into LDS
  const _Float16* qrow = qf + ((size_t)bh * E_ + qbase) * DK_;
  v16h a0 = load_a16(qrow, DK_, lane);        // K = 0..31 of dk=64
  v16h a1 = load_a16(qrow + 32, DK_, lane);   // K = 32..63
  int nbase0 = wave * 256;                    // this wave's 256-key strip
#pragma unroll 4
  for (int nt = 0; nt < 16; ++nt) {
    int nbase = nbase0 + nt * 16;
    const _Float16* krow = kf + ((size_t)bh * E_ + nbase) * DK_;
    v16h b0 = load_b16(krow, DK_, lane);
    v16h b1 = load_b16(krow + 32, DK_, lane);
    v8f c = {};
    c = wmma_f16(a0, b0, c);
    c = wmma_f16(a1, b1, c);
    int n = nbase + l16;
#pragma unroll
    for (int r = 0; r < 8; ++r) {
      int qi = qbase + r + 8 * hfl;
      int dd = dist[((size_t)b * E_ + qi) * E_ + n];
      int pos = (dd < 8) ? dd : 8;            // min(dist, MAX_POS)
      float bias = qrpr[((size_t)bh * E_ + qi) * 16 + pos];
      float val = (dd <= 8) ? (c[r] + bias) : -1e9f;
      tile[(r + 8 * hfl) * LDST + n] = val;
    }
  }
  __syncthreads();

  // -------- Phase 2: row softmax stats (wave w owns rows 2w, 2w+1)
  {
    int row = 2 * wave + hfl;
    float* rp = tile + row * LDST;
    float mx = -1e30f;
#pragma unroll 8
    for (int j = 0; j < 128; ++j) mx = fmaxf(mx, rp[l16 + 16 * j]);
#pragma unroll
    for (int o = 8; o >= 1; o >>= 1) mx = fmaxf(mx, __shfl_xor(mx, o, 32));
    float sum = 0.f;
#pragma unroll 8
    for (int j = 0; j < 128; ++j) {
      float ev = __expf(rp[l16 + 16 * j] - mx);
      rp[l16 + 16 * j] = ev;                  // overwrite logits with exp()
      sum += ev;
    }
#pragma unroll
    for (int o = 8; o >= 1; o >>= 1) sum += __shfl_xor(sum, o, 32);
    if (l16 == 0) inv_s[row] = 1.0f / sum;
  }
  __syncthreads();

  if (wave < 4) {
    // -------- Phase 3a: out[16 x 16] = softmax(S) @ V for d-tile = wave
    const _Float16* vbase = vT + ((size_t)bh * DK_ + wave * 16) * E_;
    v8f c = {};
    for (int kb = 0; kb < E_; kb += 32) {
      v16h af = load_a32(tile + kb, LDST, lane);        // exp() values, LDS
      v16h bf = load_b16(vbase + kb, E_, lane);
      c = wmma_f16(af, bf, c);
    }
#pragma unroll
    for (int r = 0; r < 8; ++r) {
      int rr = r + 8 * hfl;
      int e = qbase + rr;
      float val = c[r] * inv_s[rr];                     // fold 1/sum here
      outcat[((size_t)b * E_ + e) * HD_ + h * DK_ + wave * 16 + l16] = (_Float16)val;
    }
  } else {
    // -------- Phase 3b: stream normalized `a` slab to HBM (coalesced)
    int t = tid - 128;                                  // 0..127
    for (int i = 0; i < 256; ++i) {
      int idx = i * 128 + t;                            // 16*2048 elements
      int row = idx >> 11, col = idx & 2047;
      aout[((size_t)bh * E_ + qbase + row) * E_ + col] =
          tile[row * LDST + col] * inv_s[row];
    }
  }
}

// ---- Kernel 6: x_out[b,n,e] = (outcat @ w_fc^T)[b,e,n] + x[b,n,e] ----------

__global__ void fc_kernel(const _Float16* __restrict__ outcat, const _Float16* __restrict__ wfc,
                          const float* __restrict__ x, float* __restrict__ xout) {
  int lane = threadIdx.x & 31, wave = threadIdx.x >> 5;
  int mbase = blockIdx.y * 128 + wave * 16;   // row in [B*E]
  int nbase = blockIdx.x * 16;                // out feature in D
  v8f c = {};
#pragma unroll
  for (int kb = 0; kb < HD_; kb += 32) {
    v16h af = load_a16(outcat + (size_t)mbase * HD_ + kb, HD_, lane);
    v16h bf = load_b16(wfc + (size_t)nbase * HD_ + kb, HD_, lane);
    c = wmma_f16(af, bf, c);
  }
  int l16 = lane & 15, hfl = lane >> 4;
  int b = mbase / E_;
  int n = nbase + l16;
#pragma unroll
  for (int r = 0; r < 8; ++r) {
    int e = mbase - b * E_ + r + 8 * hfl;
    size_t o = ((size_t)b * D_ + n) * E_ + e;
    xout[o] = c[r] + x[o];                    // residual add, transposed store
  }
}

// ---------------------------------------------------------------------------

extern "C" void kernel_launch(void* const* d_in, const int* in_sizes, int n_in,
                              void* d_out, int out_size, void* d_ws, size_t ws_size,
                              hipStream_t stream) {
  (void)in_sizes; (void)n_in; (void)out_size; (void)ws_size;
  const float* x     = (const float*)d_in[0];   // [B,D,E]
  const int*   dist  = (const int*)d_in[1];     // [B,E,E]
  const float* w_qs  = (const float*)d_in[2];   // [256,256]
  const float* w_ks  = (const float*)d_in[3];
  const float* w_vs  = (const float*)d_in[4];
  const float* w_fc  = (const float*)d_in[5];
  const float* w_rpr = (const float*)d_in[6];   // [9,64]
  const float* ln_g  = (const float*)d_in[7];   // [256]
  const float* ln_b  = (const float*)d_in[8];   // [256]

  // workspace carve-up (halves): 6 x 1M-element f16 tensors + weights + qrpr
  const size_t M1 = (size_t)1 << 20;            // B*E*D == B*H*E*DK == 1M
  _Float16* s_h  = (_Float16*)d_ws;
  _Float16* qn_h = s_h  + M1;
  _Float16* qf   = qn_h + M1;                   // [B,H,E,DK], pre-scaled
  _Float16* kf   = qf   + M1;                   // [B,H,E,DK]
  _Float16* vT   = kf   + M1;                   // [B,H,DK,E]
  _Float16* outc = vT   + M1;                   // [B,E,H*DK]
  _Float16* w_h  = outc + M1;                   // wq|wk|wv|wfc, 4*65536 halves
  float*    qrpr = (float*)(w_h + 4 * 65536);   // [B,H,E,16]

  float* xout = (float*)d_out;                  // [B,D,E]
  float* aout = xout + (size_t)B_ * D_ * E_;    // [B,H,E,E]

  ln_kernel        <<<dim3(B_ * E_ / 32), 256, 0, stream>>>(x, ln_g, ln_b, s_h, qn_h);
  wcvt_kernel      <<<dim3(4 * 65536 / 256), 256, 0, stream>>>(w_qs, w_ks, w_vs, w_fc, w_h);
  proj_kernel      <<<dim3(HD_ / 16, B_ * E_ / 128, 3), 256, 0, stream>>>(s_h, qn_h, w_h, qf, kf, vT);
  qrpr_kernel      <<<dim3(B_ * H_ * E_ * 16 / 256), 256, 0, stream>>>(qf, w_rpr, qrpr);
  attn_fused_kernel<<<dim3(E_ / 16, B_ * H_), 256, 0, stream>>>(qf, kf, qrpr, dist, vT, aout, outc);
  fc_kernel        <<<dim3(D_ / 16, B_ * E_ / 128), 256, 0, stream>>>(outc, w_h + 3 * 65536, x, xout);
}
